// MixedCausalAttention_42786464203270
// MI455X (gfx1250) — matrix operand order repaired and checked by reading, not compiled, not run on hardware
//
#include <hip/hip_runtime.h>

typedef _Float16 v16h __attribute__((ext_vector_type(16)));
typedef _Float16 v8h  __attribute__((ext_vector_type(8)));
typedef _Float16 v4h  __attribute__((ext_vector_type(4)));
typedef float    v8f  __attribute__((ext_vector_type(8)));
typedef int      v4i  __attribute__((ext_vector_type(4)));

#define DEVFN __device__ __forceinline__

constexpr int B_  = 2;
constexpr int S_  = 2048;
constexpr int D_  = 1024;
constexpr int H_  = 16;
constexpr int HD_ = 64;
constexpr int NS_ = 8;
constexpr int BS_ = B_ * S_;    // 4096 rows
constexpr int N3D = 3 * D_;     // 3072

// ---------------------------------------------------------------------------
// CDNA5 async global->LDS copy (ASYNCcnt path), guarded so we always compile.
// Toolchain signature (probe-confirmed): (global v4i*, local v4i*, imm, cpol).
// ---------------------------------------------------------------------------
#if defined(__HIP_DEVICE_COMPILE__) && defined(__has_builtin)
#if __has_builtin(__builtin_amdgcn_global_load_async_to_lds_b128) && \
    __has_builtin(__builtin_amdgcn_s_wait_asynccnt)
#define USE_ASYNC_LDS 1
#endif
#endif
#ifndef USE_ASYNC_LDS
#define USE_ASYNC_LDS 0
#endif

#if USE_ASYNC_LDS
typedef __attribute__((address_space(1))) v4i* gv4i_p;  // global
typedef __attribute__((address_space(3))) v4i* lv4i_p;  // LDS
#endif

DEVFN void copy16_g2l(const _Float16* __restrict__ g, _Float16* __restrict__ l) {
#if USE_ASYNC_LDS
  __builtin_amdgcn_global_load_async_to_lds_b128((gv4i_p)g, (lv4i_p)l, 0, 0);
#else
  *(v8h*)l = *(const v8h*)g;
#endif
}

DEVFN void wait_async_copies() {
#if USE_ASYNC_LDS
  __builtin_amdgcn_s_wait_asynccnt(0);
#endif
}

DEVFN void lds_fence_wave() {
#if defined(__gfx1250__)
  asm volatile("s_wait_dscnt 0" ::: "memory");
#endif
}

DEVFN v16h mk16(v8h lo, v8h hi) {
  v16h r;
#pragma unroll
  for (int i = 0; i < 8; ++i) { r[i] = lo[i]; r[i + 8] = hi[i]; }
  return r;
}

DEVFN v8f wmma_f16(v16h a, v16h b, v8f c) {
  // 8 args: (neg_a, A, neg_b, B, c_mod, C, reuse_a, reuse_b)
  return __builtin_amdgcn_wmma_f32_16x16x32_f16(false, a, false, b, (short)0, c,
                                                false, false);
}

// ---------------------------------------------------------------------------
// fp32 -> f16 convert (vectorized x4)
// ---------------------------------------------------------------------------
__global__ void k_cvt_f16(const float* __restrict__ in, _Float16* __restrict__ out,
                          int n4) {
  int i = blockIdx.x * blockDim.x + threadIdx.x;
  if (i >= n4) return;
  float4 f = ((const float4*)in)[i];
  v4h h; h[0] = (_Float16)f.x; h[1] = (_Float16)f.y;
  h[2] = (_Float16)f.z; h[3] = (_Float16)f.w;
  ((v4h*)out)[i] = h;
}

// ---------------------------------------------------------------------------
// fp32 [K][N] -> f16 [N][K] transpose (LDS-tiled, 32x32, 256 threads)
// One-time cost so GEMM B tiles become K-contiguous async b128 DMAs.
// ---------------------------------------------------------------------------
__global__ __launch_bounds__(256) void k_cvt_transpose(
    const float* __restrict__ in, _Float16* __restrict__ out, int K, int N) {
  __shared__ float tile[32][33];
  const int k0 = blockIdx.y * 32, n0 = blockIdx.x * 32;
  const int tx = threadIdx.x & 31, ty = threadIdx.x >> 5;  // 32 x 8
#pragma unroll
  for (int i = 0; i < 32; i += 8)
    tile[ty + i][tx] = in[(size_t)(k0 + ty + i) * N + n0 + tx];
  __syncthreads();
#pragma unroll
  for (int i = 0; i < 32; i += 8)
    out[(size_t)(n0 + ty + i) * K + k0 + tx] = (_Float16)tile[tx][ty + i];
}

// ---------------------------------------------------------------------------
// Tiled WMMA GEMM:  C[M,N] = A[M,K](f16) * Bt[N,K](f16, pre-transposed) + bias
// Block tile 128x128, k-step 32, 256 threads = 8 waves, wave tile 32x64.
// Both tiles are async global->LDS b128; 2-stage pipeline, 1 barrier/iter:
//   [wait asynccnt 0][barrier][issue next tile][compute current]
// ---------------------------------------------------------------------------
__global__ __launch_bounds__(256) void k_gemm_f16(
    const _Float16* __restrict__ A, const _Float16* __restrict__ Bt,
    const float* __restrict__ bias, float* __restrict__ C, int M, int N, int K) {
  __shared__ __align__(16) _Float16 As[2][128 * 32];  // [row][k]
  __shared__ __align__(16) _Float16 Bs[2][128 * 32];  // [n][k]

  const int tid  = threadIdx.x;
  const int lane = tid & 31;
  const int wave = tid >> 5;
  const int l16  = lane & 15;
  const int lh   = lane >> 4;   // half-group 0/1
  const int wm   = wave >> 1;   // 0..3 -> 32-row strip
  const int wn   = wave & 1;    // 0..1 -> 64-col strip

  const int rowBase = blockIdx.y * 128;
  const int colBase = blockIdx.x * 128;

  auto stage = [&](int kt, int buf) {
    const int k0 = kt * 32;
#pragma unroll
    for (int p = 0; p < 2; ++p) {
      int cid = tid + p * 256;            // 0..511, 16B chunks of 128x32
      int r = cid >> 2, koff = (cid & 3) * 8;
      copy16_g2l(A + (size_t)(rowBase + r) * K + k0 + koff,
                 &As[buf][r * 32 + koff]);
      copy16_g2l(Bt + (size_t)(colBase + r) * K + k0 + koff,
                 &Bs[buf][r * 32 + koff]);
    }
  };

  v8f acc[2][4];
#pragma unroll
  for (int mi = 0; mi < 2; ++mi)
#pragma unroll
    for (int ni = 0; ni < 4; ++ni) acc[mi][ni] = v8f{};

  const int nk = K / 32;
  stage(0, 0);
  for (int kt = 0; kt < nk; ++kt) {
    const int buf = kt & 1;
    wait_async_copies();   // this wave's copies for tile kt are done
    __syncthreads();       // everyone's copies for tile kt are done
    if (kt + 1 < nk) stage(kt + 1, buf ^ 1);  // DMA overlaps compute below

    const _Float16* as = As[buf];
    const _Float16* bs = Bs[buf];
    v16h afr[2], bfr[4];
#pragma unroll
    for (int mi = 0; mi < 2; ++mi) {
      const _Float16* ap = as + (wm * 32 + mi * 16 + l16) * 32 + lh * 8;
      afr[mi] = mk16(*(const v8h*)ap, *(const v8h*)(ap + 16));
    }
#pragma unroll
    for (int ni = 0; ni < 4; ++ni) {
      const _Float16* bp = bs + (wn * 64 + ni * 16 + l16) * 32 + lh * 8;
      bfr[ni] = mk16(*(const v8h*)bp, *(const v8h*)(bp + 16));
    }
#pragma unroll
    for (int mi = 0; mi < 2; ++mi)
#pragma unroll
      for (int ni = 0; ni < 4; ++ni)
        acc[mi][ni] = wmma_f16(afr[mi], bfr[ni], acc[mi][ni]);
  }

  // C/D layout: M = v + 8*(lane>>4), N = lane&15
#pragma unroll
  for (int mi = 0; mi < 2; ++mi) {
#pragma unroll
    for (int ni = 0; ni < 4; ++ni) {
      int col = colBase + wn * 64 + ni * 16 + l16;
      float bv = bias[col];
#pragma unroll
      for (int v = 0; v < 8; ++v) {
        int row = rowBase + wm * 32 + mi * 16 + v + lh * 8;
        C[(size_t)row * N + col] = acc[mi][ni][v] + bv;
      }
    }
  }
}

// ---------------------------------------------------------------------------
// Per-position override rows: qkv[b, pos+n, proj] = tokens @ W*_ns[n] + b*_ns[n]
// seq_token_count read on-device (no host sync under graph capture).
// ---------------------------------------------------------------------------
__global__ __launch_bounds__(256) void k_ns_override(
    const float* __restrict__ tokens, const int* __restrict__ seq_cnt,
    const float* __restrict__ Wq, const float* __restrict__ bq,
    const float* __restrict__ Wk, const float* __restrict__ bk,
    const float* __restrict__ Wv, const float* __restrict__ bv,
    float* __restrict__ qkv) {
  __shared__ float tk[D_];
  const int p = blockIdx.x, n = blockIdx.y, b = blockIdx.z;
  const int pos = *seq_cnt;
  const int s = pos + n;
  if (s < 0 || s >= S_) return;  // n >= n_eff
  const float* W  = (p == 0 ? Wq : (p == 1 ? Wk : Wv)) + (size_t)n * D_ * D_;
  const float* bb = (p == 0 ? bq : (p == 1 ? bk : bv)) + (size_t)n * D_;
  const float* tok = tokens + ((size_t)b * S_ + s) * D_;
  for (int i = threadIdx.x; i < D_; i += 256) tk[i] = tok[i];
  __syncthreads();

  const int e0 = threadIdx.x * 4;
  float a0 = 0.f, a1 = 0.f, a2 = 0.f, a3 = 0.f;
  for (int d = 0; d < D_; ++d) {
    float t = tk[d];
    float4 w = *(const float4*)(W + (size_t)d * D_ + e0);
    a0 += t * w.x; a1 += t * w.y; a2 += t * w.z; a3 += t * w.w;
  }
  float* out = qkv + ((size_t)b * S_ + s) * N3D + p * D_ + e0;
  out[0] = a0 + bb[e0 + 0];
  out[1] = a1 + bb[e0 + 1];
  out[2] = a2 + bb[e0 + 2];
  out[3] = a3 + bb[e0 + 3];
}

// ---------------------------------------------------------------------------
// Split fp32 qkv [4096,3072]:
//   q,k -> f16 [B*S][D] (row-major, d-contiguous per head)
//   v   -> f16 [B][H][HD][S] (d-major) so attention PV B-tiles are s-contiguous
// ---------------------------------------------------------------------------
__global__ void k_split_qkv(const float* __restrict__ qkv,
                            _Float16* __restrict__ q, _Float16* __restrict__ k,
                            _Float16* __restrict__ vt) {
  int i = blockIdx.x * blockDim.x + threadIdx.x;
  if (i >= BS_ * N3D / 4) return;
  int idx = i * 4;
  int row = idx / N3D, c = idx - row * N3D;
  int part = c >> 10, cp = c & 1023;
  float4 f = *(const float4*)(qkv + idx);
  if (part < 2) {
    _Float16* o = (part == 0 ? q : k) + (size_t)row * D_ + cp;
    v4h h; h[0] = (_Float16)f.x; h[1] = (_Float16)f.y;
    h[2] = (_Float16)f.z; h[3] = (_Float16)f.w;
    *(v4h*)o = h;
  } else {
    int b = row >> 11, s = row & (S_ - 1);
    int hh = cp >> 6, d0 = cp & 63;
    _Float16* vo = vt + ((size_t)(b * H_ + hh) * HD_ + d0) * S_ + s;
    vo[0 * S_] = (_Float16)f.x;
    vo[1 * S_] = (_Float16)f.y;
    vo[2 * S_] = (_Float16)f.z;
    vo[3 * S_] = (_Float16)f.w;
  }
}

// ---------------------------------------------------------------------------
// Flash attention (causal + padding mask), f16 WMMA, fp32 accumulators.
// grid = (S/128, H, B); 256 threads = 8 waves; wave w owns q rows [w*16, +16).
// K blocks of 64; K and V tiles both pure async b128 DMA, double-buffered.
// ---------------------------------------------------------------------------
__global__ __launch_bounds__(256) void k_attn(
    const _Float16* __restrict__ Qg, const _Float16* __restrict__ Kg,
    const _Float16* __restrict__ Vtg, const unsigned char* __restrict__ pad,
    _Float16* __restrict__ ctx) {
  constexpr int KB = 64;  // k-block
  __shared__ __align__(16) _Float16 Ks[2][KB * HD_];   // [seq][d]
  __shared__ __align__(16) _Float16 Vs[2][HD_ * KB];   // [d][seq]
  __shared__ __align__(16) _Float16 Ps[8 * 16 * KB];   // per-wave P tile

  const int tid = threadIdx.x, lane = tid & 31, wave = tid >> 5;
  const int l16 = lane & 15, lh = lane >> 4;
  const int b = blockIdx.z, h = blockIdx.y;
  const int qRow0 = blockIdx.x * 128 + wave * 16;
  const size_t headOff = (size_t)h * HD_;
  const _Float16* Vt = Vtg + (size_t)(b * H_ + h) * HD_ * S_;  // [d][s]

  auto stage = [&](int kb, int buf) {
    const int kBase = kb * KB;
#pragma unroll
    for (int p = 0; p < 2; ++p) {
      int cid = tid + p * 256;          // 0..511, 16B chunks of 64x64
      int r = cid >> 3, coff = (cid & 7) * 8;
      copy16_g2l(Kg + (size_t)(b * S_ + kBase + r) * D_ + headOff + coff,
                 &Ks[buf][r * HD_ + coff]);
      copy16_g2l(Vt + (size_t)r * S_ + kBase + coff,
                 &Vs[buf][r * KB + coff]);
    }
  };

  // Q fragments (16x64 = two K=32 chunks), A layout: row=l16, K chunks lh*8/+16
  v16h qa[2];
#pragma unroll
  for (int kk = 0; kk < 2; ++kk) {
    const _Float16* qp =
        Qg + (size_t)(b * S_ + qRow0 + l16) * D_ + headOff + kk * 32 + lh * 8;
    qa[kk] = mk16(*(const v8h*)qp, *(const v8h*)(qp + 16));
  }

  v8f o[4];
#pragma unroll
  for (int nd = 0; nd < 4; ++nd) o[nd] = v8f{};
  float mrow[8], lrow[8];
#pragma unroll
  for (int v = 0; v < 8; ++v) { mrow[v] = -1.0e30f; lrow[v] = 0.f; }

  const float scale = 0.125f;  // 1/sqrt(64)
  const int nkb = (blockIdx.x * 128 + 127) / KB + 1;  // causal block count
  _Float16* pw = Ps + wave * 16 * KB;

  stage(0, 0);
  for (int kb = 0; kb < nkb; ++kb) {
    const int buf = kb & 1;
    const int kBase = kb * KB;
    wait_async_copies();
    __syncthreads();
    if (kb + 1 < nkb) stage(kb + 1, buf ^ 1);  // overlap next K/V DMA

    // S = Q * K^T : B frag = K[seqcol][d-chunk], contraction over d
    v8f sc[4];
#pragma unroll
    for (int ni = 0; ni < 4; ++ni) {
      sc[ni] = v8f{};
#pragma unroll
      for (int kk = 0; kk < 2; ++kk) {
        const _Float16* bp = Ks[buf] + (ni * 16 + l16) * HD_ + kk * 32 + lh * 8;
        v16h bf = mk16(*(const v8h*)bp, *(const v8h*)(bp + 16));
        sc[ni] = wmma_f16(qa[kk], bf, sc[ni]);
      }
    }

    // scale + causal/padding mask + per-row max
    float rmax[8];
#pragma unroll
    for (int v = 0; v < 8; ++v) rmax[v] = -1.0e30f;
#pragma unroll
    for (int ni = 0; ni < 4; ++ni) {
      const int kIdx = kBase + ni * 16 + l16;
      const bool padOk = pad[b * S_ + kIdx] != 0;
#pragma unroll
      for (int v = 0; v < 8; ++v) {
        const int qr = qRow0 + v + lh * 8;
        float sv = (kIdx <= qr && padOk) ? sc[ni][v] * scale : -1.0e9f;
        sc[ni][v] = sv;
        rmax[v] = fmaxf(rmax[v], sv);
      }
    }
#pragma unroll
    for (int v = 0; v < 8; ++v) {  // reduce across the 16-lane half-group
      float r = rmax[v];
      r = fmaxf(r, __shfl_xor(r, 1, 32));
      r = fmaxf(r, __shfl_xor(r, 2, 32));
      r = fmaxf(r, __shfl_xor(r, 4, 32));
      r = fmaxf(r, __shfl_xor(r, 8, 32));
      rmax[v] = r;
    }

    float alpha[8], rsum[8];
#pragma unroll
    for (int v = 0; v < 8; ++v) {
      float mnew = fmaxf(mrow[v], rmax[v]);
      alpha[v] = __expf(mrow[v] - mnew);
      mrow[v] = mnew;
      rsum[v] = 0.f;
    }
#pragma unroll
    for (int ni = 0; ni < 4; ++ni)
#pragma unroll
      for (int v = 0; v < 8; ++v) {
        float pv = __expf(sc[ni][v] - mrow[v]);
        sc[ni][v] = pv;
        rsum[v] += pv;
      }
#pragma unroll
    for (int v = 0; v < 8; ++v) {
      float r = rsum[v];
      r += __shfl_xor(r, 1, 32);
      r += __shfl_xor(r, 2, 32);
      r += __shfl_xor(r, 4, 32);
      r += __shfl_xor(r, 8, 32);
      lrow[v] = lrow[v] * alpha[v] + r;
    }
#pragma unroll
    for (int nd = 0; nd < 4; ++nd)
#pragma unroll
      for (int v = 0; v < 8; ++v) o[nd][v] *= alpha[v];

    // P: C-layout f32 -> per-wave LDS -> A-layout f16 (wave-private, in-order DS)
#pragma unroll
    for (int ni = 0; ni < 4; ++ni)
#pragma unroll
      for (int v = 0; v < 8; ++v)
        pw[(v + lh * 8) * KB + ni * 16 + l16] = (_Float16)sc[ni][v];
    lds_fence_wave();

    // O += P * V : B frag from d-major V tile (s-contiguous)
#pragma unroll
    for (int kk = 0; kk < 2; ++kk) {
      const _Float16* pp = pw + l16 * KB + kk * 32 + lh * 8;
      v16h pf = mk16(*(const v8h*)pp, *(const v8h*)(pp + 16));
#pragma unroll
      for (int nd = 0; nd < 4; ++nd) {
        const _Float16* vp = Vs[buf] + (nd * 16 + l16) * KB + kk * 32 + lh * 8;
        v16h vf = mk16(*(const v8h*)vp, *(const v8h*)(vp + 16));
        o[nd] = wmma_f16(pf, vf, o[nd]);
      }
    }
  }

  // normalize, emit ctx as f16 feeding the output projection GEMM
#pragma unroll
  for (int v = 0; v < 8; ++v) lrow[v] = 1.0f / lrow[v];
#pragma unroll
  for (int nd = 0; nd < 4; ++nd)
#pragma unroll
    for (int v = 0; v < 8; ++v) {
      int row = qRow0 + v + lh * 8;
      ctx[(size_t)(b * S_ + row) * D_ + headOff + nd * 16 + l16] =
          (_Float16)(o[nd][v] * lrow[v]);
    }
}

// ---------------------------------------------------------------------------
// Host-side orchestration
// ---------------------------------------------------------------------------
extern "C" void kernel_launch(void* const* d_in, const int* in_sizes, int n_in,
                              void* d_out, int out_size, void* d_ws,
                              size_t ws_size, hipStream_t stream) {
  (void)in_sizes; (void)n_in; (void)out_size; (void)ws_size;
  const float*         tokens = (const float*)d_in[0];
  const unsigned char* pad    = (const unsigned char*)d_in[1];
  const int*           seqcnt = (const int*)d_in[2];
  const float*         W_qkv  = (const float*)d_in[3];
  const float*         b_qkv  = (const float*)d_in[4];
  const float*         Wq_ns  = (const float*)d_in[5];
  const float*         bq_ns  = (const float*)d_in[6];
  const float*         Wk_ns  = (const float*)d_in[7];
  const float*         bk_ns  = (const float*)d_in[8];
  const float*         Wv_ns  = (const float*)d_in[9];
  const float*         bv_ns  = (const float*)d_in[10];
  const float*         W_out  = (const float*)d_in[11];
  const float*         b_out  = (const float*)d_in[12];
  float*               out    = (float*)d_out;

  char* ws = (char*)d_ws;
  size_t off = 0;
  auto take = [&](size_t bytes) -> char* {
    char* p = ws + off;
    off = (off + bytes + 255) & ~(size_t)255;
    return p;
  };
  _Float16* Ah   = (_Float16*)take((size_t)BS_ * D_ * 2);   // tokens f16
  _Float16* Wqt  = (_Float16*)take((size_t)D_ * N3D * 2);   // W_qkv^T f16 [3072][1024]
  _Float16* Wot  = (_Float16*)take((size_t)D_ * D_ * 2);    // W_out^T f16 [1024][1024]
  float*    qkv  = (float*)   take((size_t)BS_ * N3D * 4);  // fused qkv fp32
  _Float16* qh   = (_Float16*)take((size_t)BS_ * D_ * 2);
  _Float16* kh   = (_Float16*)take((size_t)BS_ * D_ * 2);
  _Float16* vt   = (_Float16*)take((size_t)BS_ * D_ * 2);   // V^T per head
  _Float16* ctxh = Ah;  // tokens f16 dead after QKV GEMM; reuse for ctx

  {
    int n4 = BS_ * D_ / 4;
    k_cvt_f16<<<(n4 + 255) / 256, 256, 0, stream>>>(tokens, Ah, n4);
  }
  // weights converted + transposed once -> K-contiguous async GEMM B tiles
  k_cvt_transpose<<<dim3(N3D / 32, D_ / 32), 256, 0, stream>>>(W_qkv, Wqt, D_, N3D);
  k_cvt_transpose<<<dim3(D_ / 32, D_ / 32), 256, 0, stream>>>(W_out, Wot, D_, D_);

  // qkv = tokens @ W_qkv + b_qkv   (4096 x 3072 x 1024)
  k_gemm_f16<<<dim3(N3D / 128, BS_ / 128), 256, 0, stream>>>(
      Ah, Wqt, b_qkv, qkv, BS_, N3D, D_);

  // per-position overrides (reads seq_token_count on device)
  k_ns_override<<<dim3(3, NS_, B_), 256, 0, stream>>>(
      tokens, seqcnt, Wq_ns, bq_ns, Wk_ns, bk_ns, Wv_ns, bv_ns, qkv);

  {
    int n4 = BS_ * N3D / 4;
    k_split_qkv<<<(n4 + 255) / 256, 256, 0, stream>>>(qkv, qh, kh, vt);
  }

  // flash attention
  k_attn<<<dim3(S_ / 128, H_, B_), 256, 0, stream>>>(qh, kh, vt, pad, ctxh);

  // out = ctx @ W_out + b_out   (4096 x 1024 x 1024)
  k_gemm_f16<<<dim3(D_ / 128, BS_ / 128), 256, 0, stream>>>(
      ctxh, Wot, b_out, out, BS_, D_, D_);
}
